// UnaryScorer_60988535603679
// MI455X (gfx1250) — matrix operand order repaired and checked by reading, not compiled, not run on hardware
//
#include <hip/hip_runtime.h>
#include <hip/hip_bf16.h>
#include <math.h>

// Problem constants (from reference)
#define B_   128
#define M_   1024
#define DZ   1024
#define DPHI 512
#define H1   512
#define H2   512

typedef __attribute__((ext_vector_type(16))) __bf16 v16bf;
typedef __attribute__((ext_vector_type(4)))  __bf16 v4bf;
typedef __attribute__((ext_vector_type(8)))  float  v8f;

// Branchless tanh-form GELU via hardware exp2 + rcp (no libm, no divergence).
// gelu(x) = x * sigmoid(2c(x + 0.044715 x^3)), c = sqrt(2/pi)
// 2c*log2(e) = 2.3022082
__device__ __forceinline__ float gelu_fast(float x) {
    const float x2 = x * x;
    const float t  = x * fmaf(x2, 0.044715f, 1.0f);
    const float e  = __builtin_amdgcn_exp2f(-2.3022082f * t);
    return x * __builtin_amdgcn_rcpf(1.0f + e);
}

// ---------------------------------------------------------------------------
// Phase 1: hz = g_q @ W1[:DZ]  ([128,512]);  hp = Phi @ W1[DZ:] + b1 ([1024,512])
// 8 rows per block to reuse W1 rows; 256 threads, 2 columns each.
// ---------------------------------------------------------------------------
__launch_bounds__(256)
__global__ void phase1_kernel(const float* __restrict__ g_q,
                              const float* __restrict__ Phi,
                              const float* __restrict__ W1,
                              const float* __restrict__ b1,
                              float* __restrict__ hz,
                              float* __restrict__ hp) {
    const int tid = threadIdx.x;
    const int c0 = tid, c1 = tid + 256;
    const int bid = blockIdx.x;

    if (bid < 16) {                       // hz rows [bid*8, bid*8+8)
        const int r0 = bid * 8;
        float acc0[8], acc1[8];
        #pragma unroll
        for (int r = 0; r < 8; ++r) { acc0[r] = 0.f; acc1[r] = 0.f; }
        for (int k = 0; k < DZ; ++k) {
            const float w0 = W1[k * H1 + c0];
            const float w1 = W1[k * H1 + c1];
            #pragma unroll
            for (int r = 0; r < 8; ++r) {
                const float x = g_q[(r0 + r) * DZ + k];   // uniform -> scalar load
                acc0[r] = fmaf(x, w0, acc0[r]);
                acc1[r] = fmaf(x, w1, acc1[r]);
            }
        }
        #pragma unroll
        for (int r = 0; r < 8; ++r) {
            hz[(r0 + r) * H1 + c0] = acc0[r];
            hz[(r0 + r) * H1 + c1] = acc1[r];
        }
    } else {                              // hp rows [(bid-16)*8, ...+8)
        const int r0 = (bid - 16) * 8;
        float acc0[8], acc1[8];
        #pragma unroll
        for (int r = 0; r < 8; ++r) { acc0[r] = 0.f; acc1[r] = 0.f; }
        for (int k = 0; k < DPHI; ++k) {
            const float w0 = W1[(DZ + k) * H1 + c0];
            const float w1 = W1[(DZ + k) * H1 + c1];
            #pragma unroll
            for (int r = 0; r < 8; ++r) {
                const float x = Phi[(r0 + r) * DPHI + k];
                acc0[r] = fmaf(x, w0, acc0[r]);
                acc1[r] = fmaf(x, w1, acc1[r]);
            }
        }
        const float bb0 = b1[c0], bb1 = b1[c1];
        #pragma unroll
        for (int r = 0; r < 8; ++r) {
            hp[(r0 + r) * H1 + c0] = acc0[r] + bb0;
            hp[(r0 + r) * H1 + c1] = acc1[r] + bb1;
        }
    }
}

// ---------------------------------------------------------------------------
// Pack W2 [H1, H2] fp32 -> bf16 tiles in WMMA B-operand lane layout.
// tileIdx = nt*16 + kt; within tile: lane*16 + e contiguous.
// B layout used here (k order matches A): lanes 0-15 -> K base 0, lanes 16-31 -> K base 16,
// element e -> K = base + e; N = nt*16 + (lane & 15).
// ---------------------------------------------------------------------------
__launch_bounds__(256)
__global__ void pack_w2_kernel(const float* __restrict__ W2,
                               __bf16* __restrict__ packed) {
    const int t = blockIdx.x * 256 + threadIdx.x;   // 0 .. 262143
    const int tile = t >> 9;
    const int rem  = t & 511;
    const int lane = rem >> 4;
    const int e    = rem & 15;
    const int nt = tile >> 4;
    const int kt = tile & 15;
    const int k = kt * 32 + ((lane < 16) ? 0 : 16) + e;
    const int n = nt * 16 + (lane & 15);
    packed[t] = (__bf16)W2[k * H2 + n];
}

// ---------------------------------------------------------------------------
// Fused main kernel: per block, 32 rows x all 512 columns.
// 256 threads = 8 waves; wave w owns 64 columns (4 N-tiles), both row-tiles.
// Per K-step: build two 16x32 bf16 A-tiles in LDS (WMMA A lane layout),
// then 8 v_wmma_f32_16x16x32_bf16.
// Epilogue: gelu(acc + b2) * W3, reduce over N (lane-xor + LDS), + b3.
// ---------------------------------------------------------------------------
__launch_bounds__(256)
__global__ void fused_kernel(const float* __restrict__ hz,
                             const float* __restrict__ hp,
                             const __bf16* __restrict__ pW2,
                             const float* __restrict__ b2,
                             const float* __restrict__ W3,
                             const float* __restrict__ b3,
                             float* __restrict__ out) {
    const int tid  = threadIdx.x;
    const int lane = tid & 31;
    const int w    = tid >> 5;           // wave id 0..7

    const int r0 = blockIdx.x * 32;      // flat row = b*M + m
    const int b  = r0 >> 10;             // / M_
    const int m0 = r0 & 1023;            // % M_

    __shared__ __align__(32) __bf16 aLds[1024];   // 2 tiles x 512 bf16
    __shared__ float part[256];                   // 8 waves x 32 rows

    const v8f vzero = {0.f, 0.f, 0.f, 0.f, 0.f, 0.f, 0.f, 0.f};
    v8f acc[8];
    #pragma unroll
    for (int i = 0; i < 8; ++i) acc[i] = vzero;

    // --- A generation indexing for this thread ---
    const int p   = tid >> 2;            // (row-tile, a-lane) pair 0..63
    const int eo  = (tid & 3) * 4;       // starting element (4 per thread)
    const int art = p >> 5;              // row tile 0/1
    const int alane = p & 31;
    const int amrow = art * 16 + (alane & 15);
    const float* hzrow = hz + b * H1;
    const float* hprow = hp + (m0 + amrow) * H1;
    // A layout: lane<16 -> K offset 0, lane>=16 -> +16; element e -> K = base + e
    const int koffbase = ((alane < 16) ? 0 : 16) + eo;
    __bf16* aStore = &aLds[p * 16 + eo];

    const __bf16* bBase = pW2 + (size_t)(w * 64) * 512 + lane * 16;

    for (int kt = 0; kt < 16; ++kt) {
        // Build A tiles (fp32 add + branchless gelu -> bf16)
        const int K0 = kt * 32 + koffbase;
        const float4 z4 = *(const float4*)(hzrow + K0);
        const float4 p4 = *(const float4*)(hprow + K0);
        v4bf a4;
        a4[0] = (__bf16)gelu_fast(z4.x + p4.x);
        a4[1] = (__bf16)gelu_fast(z4.y + p4.y);
        a4[2] = (__bf16)gelu_fast(z4.z + p4.z);
        a4[3] = (__bf16)gelu_fast(z4.w + p4.w);
        *(v4bf*)aStore = a4;
        __syncthreads();

        const v16bf a0 = *(const v16bf*)(&aLds[lane * 16]);
        const v16bf a1 = *(const v16bf*)(&aLds[512 + lane * 16]);
        const __bf16* bp = bBase + (size_t)kt * 512;
        // Load all four B tiles first so loads clause together
        const v16bf bv0 = *(const v16bf*)(bp);
        const v16bf bv1 = *(const v16bf*)(bp + (size_t)(16 * 512));
        const v16bf bv2 = *(const v16bf*)(bp + (size_t)(32 * 512));
        const v16bf bv3 = *(const v16bf*)(bp + (size_t)(48 * 512));
        acc[0] = __builtin_amdgcn_wmma_f32_16x16x32_bf16(false, a0, false, bv0, (short)0, acc[0], false, false);
        acc[4] = __builtin_amdgcn_wmma_f32_16x16x32_bf16(false, a1, false, bv0, (short)0, acc[4], false, false);
        acc[1] = __builtin_amdgcn_wmma_f32_16x16x32_bf16(false, a0, false, bv1, (short)0, acc[1], false, false);
        acc[5] = __builtin_amdgcn_wmma_f32_16x16x32_bf16(false, a1, false, bv1, (short)0, acc[5], false, false);
        acc[2] = __builtin_amdgcn_wmma_f32_16x16x32_bf16(false, a0, false, bv2, (short)0, acc[2], false, false);
        acc[6] = __builtin_amdgcn_wmma_f32_16x16x32_bf16(false, a1, false, bv2, (short)0, acc[6], false, false);
        acc[3] = __builtin_amdgcn_wmma_f32_16x16x32_bf16(false, a0, false, bv3, (short)0, acc[3], false, false);
        acc[7] = __builtin_amdgcn_wmma_f32_16x16x32_bf16(false, a1, false, bv3, (short)0, acc[7], false, false);
        __syncthreads();
    }

    // --- Epilogue: gelu(acc + b2[n]) * W3[n], reduce over N ---
    const int nbase = w * 64 + (lane & 15);
    float rs0[8], rs1[8];
    #pragma unroll
    for (int r = 0; r < 8; ++r) { rs0[r] = 0.f; rs1[r] = 0.f; }
    #pragma unroll
    for (int t = 0; t < 4; ++t) {
        const int n    = nbase + t * 16;
        const float bb = b2[n];
        const float ww = W3[n];
        #pragma unroll
        for (int r = 0; r < 8; ++r) {
            rs0[r] += gelu_fast(acc[t][r]     + bb) * ww;
            rs1[r] += gelu_fast(acc[4 + t][r] + bb) * ww;
        }
    }
    // C layout: vgpr r, lanes 0-15 = row r, lanes 16-31 = row r+8 (distinct n)
    #pragma unroll
    for (int rt = 0; rt < 2; ++rt) {
        #pragma unroll
        for (int r = 0; r < 8; ++r) {
            float s = rt ? rs1[r] : rs0[r];
            s += __shfl_xor(s, 1, 32);
            s += __shfl_xor(s, 2, 32);
            s += __shfl_xor(s, 4, 32);
            s += __shfl_xor(s, 8, 32);
            if (lane == 0)       part[w * 32 + rt * 16 + r]     = s;
            else if (lane == 16) part[w * 32 + rt * 16 + 8 + r] = s;
        }
    }
    __syncthreads();
    if (tid < 32) {
        float s = b3[0];
        #pragma unroll
        for (int ww = 0; ww < 8; ++ww) s += part[ww * 32 + tid];
        out[r0 + tid] = s;
    }
}

// ---------------------------------------------------------------------------
// Launch
// ---------------------------------------------------------------------------
extern "C" void kernel_launch(void* const* d_in, const int* in_sizes, int n_in,
                              void* d_out, int out_size, void* d_ws, size_t ws_size,
                              hipStream_t stream) {
    const float* g_q = (const float*)d_in[0];
    const float* Phi = (const float*)d_in[1];
    const float* W1  = (const float*)d_in[2];
    const float* b1  = (const float*)d_in[3];
    const float* W2  = (const float*)d_in[4];
    const float* b2  = (const float*)d_in[5];
    const float* W3  = (const float*)d_in[6];
    const float* b3  = (const float*)d_in[7];
    float* out = (float*)d_out;

    char* ws = (char*)d_ws;
    float*  hz  = (float*)(ws);                 // 128*512*4   = 256 KB
    float*  hp  = (float*)(ws + 262144);        // 1024*512*4  = 2 MB
    __bf16* pW2 = (__bf16*)(ws + 2359296);      // 512*512*2   = 512 KB

    phase1_kernel<<<144, 256, 0, stream>>>(g_q, Phi, W1, b1, hz, hp);
    pack_w2_kernel<<<1024, 256, 0, stream>>>(W2, pW2);
    fused_kernel<<<(B_ * M_) / 32, 256, 0, stream>>>(hz, hp, pW2, b2, W3, b3, out);
}
